// DevLayer_12627203850761
// MI455X (gfx1250) — matrix operand order repaired and checked by reading, not compiled, not run on hardware
//
#include <hip/hip_runtime.h>
#include <hip/hip_bf16.h>

// ---------------------------------------------------------------------------
// Problem constants (from reference): B=8, S=8192, D=512, NB=16
// ---------------------------------------------------------------------------
#define SB   8
#define SS   8192
#define DD   512
#define DD2  1024
#define NBLK 16
#define BS   (SB * SS)          // 65536 rows
#define ROWS 32                 // rows per workgroup in fused block kernels
#define SCH  32                 // S-chunks for two-stage mean reduction

typedef __bf16 bf16_t;
typedef bf16_t bf16x16 __attribute__((ext_vector_type(16)));
typedef bf16_t bf16x8  __attribute__((ext_vector_type(8)));
typedef float  f32x8   __attribute__((ext_vector_type(8)));

#define F32X8_ZERO ((f32x8){0.f,0.f,0.f,0.f,0.f,0.f,0.f,0.f})

// ---------------------------------------------------------------------------
// WMMA operand loaders (CDNA5 wave32 16x16x32 bf16 layouts, ISA 7.12.2)
//   A 16x32 (MxK): lane L -> row M=L&15; kb=(L>>4)*8;
//     elems 0..7  = K[kb..kb+7], elems 8..15 = K[16+kb..16+kb+7]
//   B 32x16 (KxN): lane L -> col N=L&15; kb=(L>>4)*8;
//     elems 0..7  = K[kb..kb+7], elems 8..15 = K[16+kb..16+kb+7]
//     (we store W^T[n][k], so B(k,n) = WT[n][k] contiguous in k)
// ---------------------------------------------------------------------------
__device__ __forceinline__ bf16x16 bf_combine(bf16x8 lo, bf16x8 hi) {
  bf16x16 r;
#pragma unroll
  for (int i = 0; i < 8; ++i) { r[i] = lo[i]; r[8 + i] = hi[i]; }
  return r;
}

// base points at [row_tile_start][k0] in a row-major bf16 buffer (LDS)
__device__ __forceinline__ bf16x16 load_A(const bf16_t* base, int stride, int lane) {
  const int row = lane & 15;
  const int kb  = (lane >> 4) * 8;
  const bf16_t* p = base + row * stride + kb;
  bf16x8 lo = *(const bf16x8*)(p);
  bf16x8 hi = *(const bf16x8*)(p + 16);
  return bf_combine(lo, hi);
}

// wt = W^T [Ntot][Kdim] bf16 in global memory; prefetch next k-chunk
__device__ __forceinline__ bf16x16 load_B(const bf16_t* wt, int Kdim, int n0, int k0, int lane) {
  const int n  = n0 + (lane & 15);
  const int kb = (lane >> 4) * 8;
  const bf16_t* p = wt + (size_t)n * Kdim + (k0 + kb);
  bf16x8 lo = *(const bf16x8*)(p);
  bf16x8 hi = *(const bf16x8*)(p + 16);
  __builtin_prefetch(p + 32, 0, 1);        // global_prefetch_b8, next k-chunk
  return bf_combine(lo, hi);
}

#define WMMA_BF16(acc, a, b) \
  __builtin_amdgcn_wmma_f32_16x16x32_bf16(false, (a), false, (b), (short)0, (acc), false, false)

// ---------------------------------------------------------------------------
// Async global -> LDS staging of ROWS x DD f32 (64 KB) via ASYNCcnt path.
// Each lane issues 16-B async copies; completion via s_wait_asynccnt.
// ---------------------------------------------------------------------------
__device__ __forceinline__ void stage_rows_async(const float* __restrict__ gsrc,
                                                 float* lds_dst, int tid) {
  const unsigned lds_base = (unsigned)(uintptr_t)lds_dst;   // LDS-relative addr
#pragma unroll
  for (int i = 0; i < (ROWS * DD) / (256 * 4); ++i) {       // 16 iters of 4 KB
    const int e = (i * 256 + tid) * 4;                      // float index
    unsigned           lv = lds_base + (unsigned)e * 4u;
    unsigned long long ga = (unsigned long long)(uintptr_t)(gsrc + e);
    asm volatile("global_load_async_to_lds_b128 %0, %1, off"
                 :: "v"(lv), "v"(ga) : "memory");
  }
  asm volatile("s_wait_asynccnt 0x0" ::: "memory");
}

// ---------------------------------------------------------------------------
// LayerNorm of ROWS x DD f32 (LDS) -> bf16 (LDS). 8 lanes per row.
// ---------------------------------------------------------------------------
__device__ __forceinline__ void ln_rows_to_bf16(const float* Xs, bf16_t* An,
                                                const float* g, const float* be,
                                                int tid) {
  const int r = tid >> 3, sub = tid & 7;
  const float* hr = Xs + r * DD;
  float s = 0.f, sq = 0.f;
  for (int j = sub; j < DD; j += 8) { float x = hr[j]; s += x; sq += x * x; }
#pragma unroll
  for (int m = 1; m < 8; m <<= 1) {
    s  += __shfl_xor(s,  m, 8);
    sq += __shfl_xor(sq, m, 8);
  }
  const float mean = s * (1.f / DD);
  const float var  = sq * (1.f / DD) - mean * mean;
  const float rstd = rsqrtf(var + 1e-5f);
  bf16_t* ar = An + r * DD;
  for (int j = sub; j < DD; j += 8)
    ar[j] = (bf16_t)((hr[j] - mean) * rstd * g[j] + be[j]);
}

// ---------------------------------------------------------------------------
// Kernel 1a: partial sums over S-chunks -> partials [2][B][SCH][D]
// 1024 blocks x 256 threads; each thread sums 256 consecutive s steps.
// ---------------------------------------------------------------------------
__global__ __launch_bounds__(256) void mean_partial_kernel(const float* __restrict__ emb_in,
                                                           const float* __restrict__ dis_in,
                                                           float* __restrict__ partials) {
  const int bx    = blockIdx.x;              // 2*8*32*2 = 1024
  const int which = bx >> 9;
  const int b     = (bx >> 6) & 7;
  const int chunk = (bx >> 1) & 31;
  const int d     = (bx & 1) * 256 + threadIdx.x;
  const float* src = which ? dis_in : emb_in;
  const float* p = src + (size_t)b * SS * DD + (size_t)chunk * (SS / SCH) * DD + d;
  float s = 0.f;
  for (int i = 0; i < SS / SCH; ++i) s += p[(size_t)i * DD];
  partials[(((size_t)which * SB + b) * SCH + chunk) * DD + d] = s;
}

// ---------------------------------------------------------------------------
// Kernel 1b: reduce partials -> means [2][B][D]
// ---------------------------------------------------------------------------
__global__ __launch_bounds__(256) void mean_reduce_kernel(const float* __restrict__ partials,
                                                          float* __restrict__ means) {
  const int bx    = blockIdx.x;              // 2*8*2 = 32
  const int which = bx >> 4;
  const int b     = (bx >> 1) & 7;
  const int d     = (bx & 1) * 256 + threadIdx.x;
  const float* p = partials + (((size_t)which * SB + b) * SCH) * DD + d;
  float s = 0.f;
#pragma unroll
  for (int c = 0; c < SCH; ++c) s += p[(size_t)c * DD];
  means[which * (SB * DD) + b * DD + d] = s * (1.f / SS);
}

// ---------------------------------------------------------------------------
// Kernel 2: sigmoid selector + top-2 for both branches (batch 0) -> sel[4]
// ---------------------------------------------------------------------------
__global__ void router_kernel(const float* __restrict__ means,
                              const float* __restrict__ esW, const float* __restrict__ esb,
                              const float* __restrict__ dsW, const float* __restrict__ dsb,
                              int* __restrict__ sel) {
  __shared__ float sc[2][NBLK];
  const int lane = threadIdx.x;
  for (int which = 0; which < 2; ++which) {
    if (lane < NBLK) {
      const float* m = means + which * (SB * DD);   // batch 0 row
      const float* W = which ? dsW : esW;           // [D, NB]
      float s        = which ? dsb[lane] : esb[lane];
      for (int k = 0; k < DD; ++k) s += m[k] * W[k * NBLK + lane];
      sc[which][lane] = 1.f / (1.f + expf(-s));
    }
  }
  __syncthreads();
  if (lane == 0) {
    for (int which = 0; which < 2; ++which) {
      int i0 = 0; float v0 = sc[which][0];
      for (int i = 1; i < NBLK; ++i) if (sc[which][i] > v0) { v0 = sc[which][i]; i0 = i; }
      int i1 = (i0 == 0) ? 1 : 0; float v1 = sc[which][i1];
      for (int i = 0; i < NBLK; ++i)
        if (i != i0 && sc[which][i] > v1) { v1 = sc[which][i]; i1 = i; }
      sel[which * 2 + 0] = i0;
      sel[which * 2 + 1] = i1;
    }
  }
}

// ---------------------------------------------------------------------------
// Kernel 3: convert the 10 selected weight matrices f32 [K][N] -> bf16 W^T [N][K]
// One 32x32 tile per workgroup via LDS transpose.
// ---------------------------------------------------------------------------
__global__ __launch_bounds__(256) void wprep_kernel(const float* __restrict__ ew1,
                                                    const float* __restrict__ ew2,
                                                    const float* __restrict__ dattn,
                                                    const float* __restrict__ dff1,
                                                    const float* __restrict__ dff2,
                                                    const int* __restrict__ sel,
                                                    bf16_t* __restrict__ wsW) {
  const int jobStart[11] = {0, 256, 512, 768, 1024, 1280, 1792, 2304, 2560, 3072, 3584};
  const int bx = blockIdx.x;
  int job = 0;
  while (job < 10 && bx >= jobStart[job + 1]) ++job;
  const int ti = bx - jobStart[job];

  const float* src; int K, N; size_t dstoff;
  switch (job) {
    case 0:  src = ew1   + (size_t)sel[0] * 262144; K = 512;  N = 512;  dstoff = 0;       break;
    case 1:  src = ew2   + (size_t)sel[0] * 262144; K = 512;  N = 512;  dstoff = 262144;  break;
    case 2:  src = ew1   + (size_t)sel[1] * 262144; K = 512;  N = 512;  dstoff = 524288;  break;
    case 3:  src = ew2   + (size_t)sel[1] * 262144; K = 512;  N = 512;  dstoff = 786432;  break;
    case 4:  src = dattn + (size_t)sel[2] * 262144; K = 512;  N = 512;  dstoff = 1048576; break;
    case 5:  src = dff1  + (size_t)sel[2] * 524288; K = 512;  N = 1024; dstoff = 1310720; break;
    case 6:  src = dff2  + (size_t)sel[2] * 524288; K = 1024; N = 512;  dstoff = 1835008; break;
    case 7:  src = dattn + (size_t)sel[3] * 262144; K = 512;  N = 512;  dstoff = 2359296; break;
    case 8:  src = dff1  + (size_t)sel[3] * 524288; K = 512;  N = 1024; dstoff = 2621440; break;
    default: src = dff2  + (size_t)sel[3] * 524288; K = 1024; N = 512;  dstoff = 3145728; break;
  }
  const int tilesK = K >> 5;
  const int k0 = (ti % tilesK) * 32;
  const int n0 = (ti / tilesK) * 32;

  __shared__ float tile[32][33];
  const int tid = threadIdx.x;
  {
    const int nl = tid & 31, kq = tid >> 5;
#pragma unroll
    for (int i = 0; i < 4; ++i) {
      const int k = kq + 8 * i;
      tile[k][nl] = src[(size_t)(k0 + k) * N + (n0 + nl)];
    }
  }
  __syncthreads();
  {
    const int kl = tid & 31, nq = tid >> 5;
    bf16_t* dst = wsW + dstoff;
#pragma unroll
    for (int i = 0; i < 4; ++i) {
      const int n = nq + 8 * i;
      dst[(size_t)(n0 + n) * K + (k0 + kl)] = (bf16_t)tile[kl][n];
    }
  }
}

// ---------------------------------------------------------------------------
// Kernel 4: fused embodied block.  h' = h + (tanh(LN(h)@W1+b1)@W2+b2)*temb*0.3
// Each wave owns n-tiles; both 16-row tiles computed per B load (B reuse x2,
// two independent WMMA accumulator chains for XDL latency hiding).
// LDS: Hs f32 [32][512] (64KB) | An bf16 [32][512] (32KB) | Ts bf16 [32][512] (32KB)
// ---------------------------------------------------------------------------
__global__ __launch_bounds__(256) void emb_block_kernel(const float* __restrict__ src,
                                                        float* __restrict__ dst,
                                                        const float* __restrict__ torsion,
                                                        const float* __restrict__ lng,
                                                        const float* __restrict__ lnb,
                                                        const float* __restrict__ b1a,
                                                        const float* __restrict__ b2a,
                                                        const bf16_t* __restrict__ w1t,
                                                        const bf16_t* __restrict__ w2t,
                                                        const int* __restrict__ sel,
                                                        int slot) {
  extern __shared__ char smem[];
  float*  Hs = (float*)smem;                                  // [32][512]
  bf16_t* An = (bf16_t*)(smem + ROWS * DD * 4);               // [32][512]
  bf16_t* Ts = (bf16_t*)(smem + ROWS * DD * 4 + ROWS * DD * 2);

  const int tid = threadIdx.x, wave = tid >> 5, lane = tid & 31;
  const long row0 = (long)blockIdx.x * ROWS;
  const int  b    = (int)(row0 >> 13);                        // row / S
  const int  idx  = sel[slot];

  stage_rows_async(src + row0 * DD, Hs, tid);
  __syncthreads();
  ln_rows_to_bf16(Hs, An, lng + (size_t)idx * DD, lnb + (size_t)idx * DD, tid);
  __syncthreads();

  // GEMM1 + tanh -> Ts
  {
    const float* b1 = b1a + (size_t)idx * DD;
    for (int nt = wave; nt < 32; nt += 8) {
      const int n0 = nt * 16;
      f32x8 acc0 = F32X8_ZERO, acc1 = F32X8_ZERO;
      for (int k0 = 0; k0 < DD; k0 += 32) {
        bf16x16 bm = load_B(w1t, DD, n0, k0, lane);
        bf16x16 a0 = load_A(An + k0,           DD, lane);
        bf16x16 a1 = load_A(An + 16 * DD + k0, DD, lane);
        acc0 = WMMA_BF16(acc0, a0, bm);
        acc1 = WMMA_BF16(acc1, a1, bm);
      }
      const int   n    = n0 + (lane & 15);
      const float bias = b1[n];
      const int   mb   = (lane >> 4) << 3;
#pragma unroll
      for (int r = 0; r < 8; ++r) {
        Ts[(mb + r) * DD + n]        = (bf16_t)tanhf(acc0[r] + bias);
        Ts[(16 + mb + r) * DD + n]   = (bf16_t)tanhf(acc1[r] + bias);
      }
    }
  }
  __syncthreads();

  // GEMM2 + gated residual -> dst
  {
    const float* b2 = b2a + (size_t)idx * DD;
    for (int nt = wave; nt < 32; nt += 8) {
      const int n0 = nt * 16;
      f32x8 acc0 = F32X8_ZERO, acc1 = F32X8_ZERO;
      for (int k0 = 0; k0 < DD; k0 += 32) {
        bf16x16 bm = load_B(w2t, DD, n0, k0, lane);
        bf16x16 a0 = load_A(Ts + k0,           DD, lane);
        bf16x16 a1 = load_A(Ts + 16 * DD + k0, DD, lane);
        acc0 = WMMA_BF16(acc0, a0, bm);
        acc1 = WMMA_BF16(acc1, a1, bm);
      }
      const int   n    = n0 + (lane & 15);
      const float bias = b2[n];
      const float temb = 1.f + torsion[b * DD + n] * 0.1f;
      const int   mb   = (lane >> 4) << 3;
#pragma unroll
      for (int r = 0; r < 8; ++r) {
        const int m0 = mb + r, m1 = 16 + mb + r;
        dst[(row0 + m0) * DD + n] = Hs[m0 * DD + n] + (acc0[r] + bias) * temb * 0.3f;
        dst[(row0 + m1) * DD + n] = Hs[m1 * DD + n] + (acc1[r] + bias) * temb * 0.3f;
      }
    }
  }
}

// ---------------------------------------------------------------------------
// Kernel 5: fused disembodied block.
// LDS: Xs f32 [32][512] (64KB) | An bf16 [32][512] (32KB) | Ts bf16 [32][1024] (64KB)
// ---------------------------------------------------------------------------
__global__ __launch_bounds__(256) void dis_block_kernel(
    const float* __restrict__ src, float* __restrict__ dst,
    const float* __restrict__ torsion, const float* __restrict__ delayed,
    const int* __restrict__ unlocked,
    const float* __restrict__ ln1g, const float* __restrict__ ln1b,
    const float* __restrict__ attnb,
    const float* __restrict__ ln2g, const float* __restrict__ ln2b,
    const float* __restrict__ ff1b, const float* __restrict__ ff2b,
    const bf16_t* __restrict__ attnW, const bf16_t* __restrict__ ff1W,
    const bf16_t* __restrict__ ff2W,
    const int* __restrict__ sel, int slot) {
  if (*unlocked == 0) return;
  extern __shared__ char smem[];
  float*  Xs = (float*)smem;                                    // [32][512] dh -> hm
  bf16_t* An = (bf16_t*)(smem + ROWS * DD * 4);                 // [32][512]
  bf16_t* Ts = (bf16_t*)(smem + ROWS * DD * 4 + ROWS * DD * 2); // [32][1024]

  const int tid = threadIdx.x, wave = tid >> 5, lane = tid & 31;
  const long row0 = (long)blockIdx.x * ROWS;
  const int  b    = (int)(row0 >> 13);
  const int  idx  = sel[2 + slot];

  stage_rows_async(src + row0 * DD, Xs, tid);
  __syncthreads();
  ln_rows_to_bf16(Xs, An, ln1g + (size_t)idx * DD, ln1b + (size_t)idx * DD, tid);
  __syncthreads();

  // attn GEMM; Xs <- hm = dh + 0.5*(acc+b+0.2*delayed)*tdis   (in place)
  {
    const float* bb = attnb + (size_t)idx * DD;
    for (int nt = wave; nt < 32; nt += 8) {
      const int n0 = nt * 16;
      f32x8 acc0 = F32X8_ZERO, acc1 = F32X8_ZERO;
      for (int k0 = 0; k0 < DD; k0 += 32) {
        bf16x16 bm = load_B(attnW, DD, n0, k0, lane);
        bf16x16 a0 = load_A(An + k0,           DD, lane);
        bf16x16 a1 = load_A(An + 16 * DD + k0, DD, lane);
        acc0 = WMMA_BF16(acc0, a0, bm);
        acc1 = WMMA_BF16(acc1, a1, bm);
      }
      const int   n    = n0 + (lane & 15);
      const float bias = bb[n];
      const float del  = delayed[b * DD + n] * 0.2f;
      const float tdis = 1.f + torsion[b * DD + n] * 0.05f;
      const int   mb   = (lane >> 4) << 3;
#pragma unroll
      for (int r = 0; r < 8; ++r) {
        const int m0 = mb + r, m1 = 16 + mb + r;
        Xs[m0 * DD + n] += 0.5f * ((acc0[r] + bias + del) * tdis);
        Xs[m1 * DD + n] += 0.5f * ((acc1[r] + bias + del) * tdis);
      }
    }
  }
  __syncthreads();
  ln_rows_to_bf16(Xs, An, ln2g + (size_t)idx * DD, ln2b + (size_t)idx * DD, tid);
  __syncthreads();

  // ff1 (512 -> 1024) + exact gelu -> Ts
  {
    const float* bb = ff1b + (size_t)idx * DD2;
    for (int nt = wave; nt < 64; nt += 8) {
      const int n0 = nt * 16;
      f32x8 acc0 = F32X8_ZERO, acc1 = F32X8_ZERO;
      for (int k0 = 0; k0 < DD; k0 += 32) {
        bf16x16 bm = load_B(ff1W, DD, n0, k0, lane);
        bf16x16 a0 = load_A(An + k0,           DD, lane);
        bf16x16 a1 = load_A(An + 16 * DD + k0, DD, lane);
        acc0 = WMMA_BF16(acc0, a0, bm);
        acc1 = WMMA_BF16(acc1, a1, bm);
      }
      const int   n    = n0 + (lane & 15);
      const float bias = bb[n];
      const int   mb   = (lane >> 4) << 3;
#pragma unroll
      for (int r = 0; r < 8; ++r) {
        const float u0 = acc0[r] + bias;
        const float u1 = acc1[r] + bias;
        Ts[(mb + r) * DD2 + n]      = (bf16_t)(0.5f * u0 * (1.f + erff(u0 * 0.70710678118f)));
        Ts[(16 + mb + r) * DD2 + n] = (bf16_t)(0.5f * u1 * (1.f + erff(u1 * 0.70710678118f)));
      }
    }
  }
  __syncthreads();

  // ff2 (1024 -> 512) + residual -> dst
  {
    const float* bb = ff2b + (size_t)idx * DD;
    for (int nt = wave; nt < 32; nt += 8) {
      const int n0 = nt * 16;
      f32x8 acc0 = F32X8_ZERO, acc1 = F32X8_ZERO;
      for (int k0 = 0; k0 < DD2; k0 += 32) {
        bf16x16 bm = load_B(ff2W, DD2, n0, k0, lane);
        bf16x16 a0 = load_A(Ts + k0,            DD2, lane);
        bf16x16 a1 = load_A(Ts + 16 * DD2 + k0, DD2, lane);
        acc0 = WMMA_BF16(acc0, a0, bm);
        acc1 = WMMA_BF16(acc1, a1, bm);
      }
      const int   n    = n0 + (lane & 15);
      const float bias = bb[n];
      const int   mb   = (lane >> 4) << 3;
#pragma unroll
      for (int r = 0; r < 8; ++r) {
        const int m0 = mb + r, m1 = 16 + mb + r;
        dst[(row0 + m0) * DD + n] = Xs[m0 * DD + n] + 0.5f * (acc0[r] + bias);
        dst[(row0 + m1) * DD + n] = Xs[m1 * DD + n] + 0.5f * (acc1[r] + bias);
      }
    }
  }
}

// ---------------------------------------------------------------------------
// Host launcher
// ---------------------------------------------------------------------------
extern "C" void kernel_launch(void* const* d_in, const int* in_sizes, int n_in,
                              void* d_out, int out_size, void* d_ws, size_t ws_size,
                              hipStream_t stream) {
  const float* emb_input  = (const float*)d_in[0];
  const float* dis_input  = (const float*)d_in[1];
  const float* torsion    = (const float*)d_in[2];
  const int*   dis_unlock = (const int*)  d_in[3];
  const float* emb_sel_W  = (const float*)d_in[5];
  const float* emb_sel_b  = (const float*)d_in[6];
  const float* emb_ln_g   = (const float*)d_in[7];
  const float* emb_ln_b   = (const float*)d_in[8];
  const float* emb_w1     = (const float*)d_in[9];
  const float* emb_b1     = (const float*)d_in[10];
  const float* emb_w2     = (const float*)d_in[11];
  const float* emb_b2     = (const float*)d_in[12];
  const float* dis_sel_W  = (const float*)d_in[13];
  const float* dis_sel_b  = (const float*)d_in[14];
  const float* dis_ln1_g  = (const float*)d_in[15];
  const float* dis_ln1_b  = (const float*)d_in[16];
  const float* dis_attn_W = (const float*)d_in[17];
  const float* dis_attn_b = (const float*)d_in[18];
  const float* dis_ln2_g  = (const float*)d_in[19];
  const float* dis_ln2_b  = (const float*)d_in[20];
  const float* dis_ff1_W  = (const float*)d_in[21];
  const float* dis_ff1_b  = (const float*)d_in[22];
  const float* dis_ff2_W  = (const float*)d_in[23];
  const float* dis_ff2_b  = (const float*)d_in[24];

  float* out_emb = (float*)d_out;
  float* out_dis = out_emb + (size_t)BS * DD;

  // workspace carve:
  //   [0, 32KB)            means   [2][B][D] f32
  //   [32KB, 32KB+256)     sel[4]  int
  //   [33024, +1MB)        mean partials [2][B][SCH][D] f32
  //   [1081600, +7MB)      bf16 W^T weights
  char*   ws       = (char*)d_ws;
  float*  means    = (float*)ws;
  int*    sel      = (int*)(ws + 32768);
  float*  partials = (float*)(ws + 33024);
  bf16_t* wsW      = (bf16_t*)(ws + 33024 + (size_t)2 * SB * SCH * DD * 4);
  const float* dis_means = means + SB * DD;

  mean_partial_kernel<<<1024, 256, 0, stream>>>(emb_input, dis_input, partials);
  mean_reduce_kernel <<<32, 256, 0, stream>>>(partials, means);
  router_kernel      <<<1, 32, 0, stream>>>(means, emb_sel_W, emb_sel_b,
                                            dis_sel_W, dis_sel_b, sel);
  wprep_kernel       <<<3584, 256, 0, stream>>>(emb_w1, emb_w2, dis_attn_W,
                                                dis_ff1_W, dis_ff2_W, sel, wsW);

  const size_t embSmem = (size_t)ROWS * DD * 4 + 2 * (size_t)ROWS * DD * 2;   // 128 KB
  const size_t disSmem = (size_t)ROWS * DD * 4 + (size_t)ROWS * DD * 2
                       + (size_t)ROWS * DD2 * 2;                              // 160 KB
  const int nblk = BS / ROWS;                                                 // 2048

  const bf16_t* we1_0 = wsW + 0;
  const bf16_t* we2_0 = wsW + 262144;
  const bf16_t* we1_1 = wsW + 524288;
  const bf16_t* we2_1 = wsW + 786432;

  emb_block_kernel<<<nblk, 256, embSmem, stream>>>(emb_input, out_emb, torsion,
      emb_ln_g, emb_ln_b, emb_b1, emb_b2, we1_0, we2_0, sel, 0);
  emb_block_kernel<<<nblk, 256, embSmem, stream>>>(out_emb, out_emb, torsion,
      emb_ln_g, emb_ln_b, emb_b1, emb_b2, we1_1, we2_1, sel, 1);

  for (int slot = 0; slot < 2; ++slot) {
    const bf16_t* attnW = wsW + 1048576 + (size_t)slot * 1310720;
    const bf16_t* ff1W  = attnW + 262144;
    const bf16_t* ff2W  = attnW + 786432;
    const float*  srcp  = (slot == 0) ? dis_input : out_dis;
    dis_block_kernel<<<nblk, 256, disSmem, stream>>>(srcp, out_dis, torsion,
        dis_means, dis_unlock,
        dis_ln1_g, dis_ln1_b, dis_attn_b, dis_ln2_g, dis_ln2_b,
        dis_ff1_b, dis_ff2_b, attnW, ff1W, ff2W, sel, slot);
  }
}
